// OrthoLinear_11510512353898
// MI455X (gfx1250) — matrix operand order repaired and checked by reading, not compiled, not run on hardware
//
#include <hip/hip_runtime.h>

#define IN_F   4096
#define OUT_F  4096
#define NTOK   4096       // 4 * 1024 tokens
#define KPR    204        // nnz per row (uniform crow ptr)

typedef __attribute__((ext_vector_type(16))) _Float16 v16h;
typedef __attribute__((ext_vector_type(8)))  _Float16 v8h;
typedef __attribute__((ext_vector_type(8)))  float    v8f;

// ---------------------------------------------------------------------------
// Kernel 1: dequantize int4 (offset-8) * row scale -> f16, stored TRANSPOSED:
// Wt[in][out]. 64x64 tile per block via LDS transpose; coalesced reads of
// packed[out][in/2] and coalesced b128 stores of Wt rows. One-time ~32MB.
// Pre-transposing W makes the GEMM's B operand K-major so the 32x16 WMMA B
// tile is staged from Wt rows by b128 LDS stores -- no scattered ds_store_b16.
// ---------------------------------------------------------------------------
#define TD 64
__global__ __launch_bounds__(256) void dequant_wt_k(const int* __restrict__ packed,
                                                    const float* __restrict__ scales,
                                                    _Float16* __restrict__ Wt) {
    __shared__ __align__(16) _Float16 tile[TD][TD + 8];
    const int t    = threadIdx.x;
    const int out0 = blockIdx.x * TD;
    const int in0  = blockIdx.y * TD;

    // read phase: row r (out), 8 packed ints -> 16 f16 of that row
    const int r  = t >> 2;
    const int cq = (t & 3) * 8;
    const float s = scales[out0 + r];
    const int* p = packed + (size_t)(out0 + r) * (IN_F / 2) + in0 / 2 + cq;
    int4 p0 = *(const int4*)p;
    int4 p1 = *(const int4*)(p + 4);
    int v8i[8] = {p0.x, p0.y, p0.z, p0.w, p1.x, p1.y, p1.z, p1.w};
    _Float16* dst = &tile[r][cq * 2];
#pragma unroll
    for (int i = 0; i < 8; ++i) {
        dst[2*i]   = (_Float16)((float)((v8i[i] & 0xF)        - 8) * s);
        dst[2*i+1] = (_Float16)((float)(((v8i[i] >> 4) & 0xF) - 8) * s);
    }
    __syncthreads();

    // write phase: in-row ir, 16 out-columns, gathered from LDS, b128 stores
    const int ir = t >> 2;
    const int jb = (t & 3) * 16;
    v8h o0, o1;
#pragma unroll
    for (int j = 0; j < 8; ++j) o0[j] = tile[jb + j][ir];
#pragma unroll
    for (int j = 0; j < 8; ++j) o1[j] = tile[jb + 8 + j][ir];
    _Float16* q = Wt + (size_t)(in0 + ir) * OUT_F + out0 + jb;
    *(v8h*)q       = o0;
    *(v8h*)(q + 8) = o1;
}

// ---------------------------------------------------------------------------
// Kernel 2: scatter-add CSR ortho values into f16 Wt (one thread per nnz).
// CAS loop handles duplicate indices; collisions are vanishingly rare.
// ---------------------------------------------------------------------------
__device__ inline void atomic_add_f16(_Float16* addr, float v) {
    unsigned long long ai = (unsigned long long)addr;
    unsigned* word = (unsigned*)(ai & ~3ULL);
    bool hi = (ai & 2) != 0;
    unsigned old = *word, assumed;
    do {
        assumed = old;
        unsigned short bits = hi ? (unsigned short)(assumed >> 16)
                                 : (unsigned short)(assumed & 0xFFFFu);
        _Float16 cur = __builtin_bit_cast(_Float16, bits);
        _Float16 sum = (_Float16)((float)cur + v);
        unsigned short nb = __builtin_bit_cast(unsigned short, sum);
        unsigned newv = hi ? ((assumed & 0x0000FFFFu) | ((unsigned)nb << 16))
                           : ((assumed & 0xFFFF0000u) | (unsigned)nb);
        old = atomicCAS(word, assumed, newv);
    } while (old != assumed);
}

__global__ __launch_bounds__(256) void csr_add_k(const _Float16* __restrict__ vals,
                                                 const int* __restrict__ idx,
                                                 _Float16* __restrict__ Wt, int nnz) {
    int j = blockIdx.x * blockDim.x + threadIdx.x;
    if (j >= nnz) return;
    int row = j / KPR;                 // out index (ortho_ptr is uniform KPR stride)
    int col = idx[j];                  // in index
    atomic_add_f16(Wt + (size_t)col * OUT_F + row, (float)vals[j]);
}

// ---------------------------------------------------------------------------
// Kernel 3: x f32 -> f16
// ---------------------------------------------------------------------------
__global__ __launch_bounds__(256) void convert_x_k(const float* __restrict__ x,
                                                   _Float16* __restrict__ xh) {
    int i = (blockIdx.x * blockDim.x + threadIdx.x) * 8;
    float4 a = *(const float4*)(x + i);
    float4 b = *(const float4*)(x + i + 4);
    v8h o = {(_Float16)a.x, (_Float16)a.y, (_Float16)a.z, (_Float16)a.w,
             (_Float16)b.x, (_Float16)b.y, (_Float16)b.z, (_Float16)b.w};
    *(v8h*)(xh + i) = o;
}

// ---------------------------------------------------------------------------
// Kernel 4: Out[N, OUT] = Xh[N, K] * Wt[K, OUT]  via v_wmma_f32_16x16x32_f16.
// Block tile 128x256, K-step 32, 8 waves; each wave owns 64x64 = 4x4 WMMA
// tiles (16 WMMA per K-step off 8 fragment loads). LDS stores and A fragment reads are b128.
// Double-buffered LDS + register prefetch of the next K tile.
// ---------------------------------------------------------------------------
#define BM 128
#define BN 256
#define BK 32
#define PK 40      // As pitch (f16) -> conflict-free A fragment reads
#define PN 264     // Bs pitch (f16), 16B-aligned rows
#define KSTEPS (IN_F / BK)

__global__ __launch_bounds__(256) void gemm_wmma_k(const _Float16* __restrict__ X,
                                                   const _Float16* __restrict__ Wt,
                                                   float* __restrict__ Out) {
    __shared__ __align__(16) _Float16 As[2][BM * PK];
    __shared__ __align__(16) _Float16 Bs[2][BK * PN];

    const int tid  = threadIdx.x;
    const int lane = tid & 31;
    const int wave = tid >> 5;
    const int wm   = wave & 1;           // M offset = wm*64
    const int wn   = wave >> 1;          // N offset = wn*64
    const int m0   = blockIdx.y * BM;
    const int n0   = blockIdx.x * BN;

    // A staging: 2 threads/row, 16 f16 each (2 x b128)
    const int alr = tid >> 1, alh = tid & 1;
    const _Float16* gA = X + (size_t)(m0 + alr) * IN_F + alh * 16;
    const unsigned  aoff = alr * PK + alh * 16;
    // B staging: 8 threads/k-row, 32 f16 each (4 x b128), rows of Wt
    const int bkr = tid >> 3, bsg = tid & 7;
    const _Float16* gB = Wt + (size_t)bkr * OUT_F + n0 + bsg * 32;
    const unsigned  boff = bkr * PN + bsg * 32;

    v8f acc[4][4] = {};
    const int fl = lane & 15;
    const int fk = (lane >> 4) * 8;

    v8h ra0, ra1, rb0, rb1, rb2, rb3;

    // preload K-step 0 into buffer 0
    ra0 = *(const v8h*)gA;        ra1 = *(const v8h*)(gA + 8);
    rb0 = *(const v8h*)gB;        rb1 = *(const v8h*)(gB + 8);
    rb2 = *(const v8h*)(gB + 16); rb3 = *(const v8h*)(gB + 24);
    *(v8h*)(&As[0][aoff])      = ra0;  *(v8h*)(&As[0][aoff + 8])  = ra1;
    *(v8h*)(&Bs[0][boff])      = rb0;  *(v8h*)(&Bs[0][boff + 8])  = rb1;
    *(v8h*)(&Bs[0][boff + 16]) = rb2;  *(v8h*)(&Bs[0][boff + 24]) = rb3;
    __syncthreads();

    for (int kt = 0; kt < KSTEPS; ++kt) {
        const int cur = kt & 1;

        // register-prefetch next K tile (hidden under the 16 WMMAs)
        if (kt + 1 < KSTEPS) {
            const _Float16* pA = gA + (size_t)(kt + 1) * BK;
            const _Float16* pB = gB + (size_t)(kt + 1) * BK * OUT_F;
            ra0 = *(const v8h*)pA;        ra1 = *(const v8h*)(pA + 8);
            rb0 = *(const v8h*)pB;        rb1 = *(const v8h*)(pB + 8);
            rb2 = *(const v8h*)(pB + 16); rb3 = *(const v8h*)(pB + 24);
            if (kt + 2 < KSTEPS) {
                __builtin_prefetch(gA + (size_t)(kt + 2) * BK, 0, 1);
                __builtin_prefetch(gB + (size_t)(kt + 2) * BK * OUT_F, 0, 1);
            }
        }

        // B fragments: lane&15 == n column; K = fk..fk+7, fk+16..fk+23 (matches A)
        v16h bf[4];
#pragma unroll
        for (int tn = 0; tn < 4; ++tn) {
            const _Float16* p = &Bs[cur][fk * PN + wn * 64 + tn * 16 + fl];
#pragma unroll
            for (int s = 0; s < 8; ++s) { bf[tn][s] = p[s * PN]; bf[tn][s + 8] = p[(s + 16) * PN]; }
        }

        // A fragments streamed; 4 WMMAs per A fragment
#pragma unroll
        for (int tm = 0; tm < 4; ++tm) {
            const _Float16* p = &As[cur][(wm * 64 + tm * 16 + fl) * PK + fk];
            v8h lo = *(const v8h*)p;         // K = fk .. fk+7
            v8h hi = *(const v8h*)(p + 16);  // K = fk+16 .. fk+23
            v16h af = __builtin_shufflevector(lo, hi, 0,1,2,3,4,5,6,7,8,9,10,11,12,13,14,15);
#pragma unroll
            for (int tn = 0; tn < 4; ++tn) {
                acc[tm][tn] = __builtin_amdgcn_wmma_f32_16x16x32_f16(
                    false, af, false, bf[tn], (short)0, acc[tm][tn], false, false);
            }
        }

        // stage next tile into the other buffer
        if (kt + 1 < KSTEPS) {
            const int nxt = cur ^ 1;
            *(v8h*)(&As[nxt][aoff])      = ra0;  *(v8h*)(&As[nxt][aoff + 8])  = ra1;
            *(v8h*)(&Bs[nxt][boff])      = rb0;  *(v8h*)(&Bs[nxt][boff + 8])  = rb1;
            *(v8h*)(&Bs[nxt][boff + 16]) = rb2;  *(v8h*)(&Bs[nxt][boff + 24]) = rb3;
        }
        __syncthreads();
    }

    // epilogue: C layout VGPR r -> M = r + 8*(lane>=16), N = lane&15
    float* ob = Out + (size_t)(m0 + wm * 64) * OUT_F + n0 + wn * 64;
    const int mo = (lane >> 4) * 8;
    const int nn = lane & 15;
#pragma unroll
    for (int tm = 0; tm < 4; ++tm)
#pragma unroll
        for (int tn = 0; tn < 4; ++tn)
#pragma unroll
            for (int r = 0; r < 8; ++r)
                ob[(size_t)(tm * 16 + mo + r) * OUT_F + tn * 16 + nn] = acc[tm][tn][r];
}

// ---------------------------------------------------------------------------
// Launch
// ---------------------------------------------------------------------------
extern "C" void kernel_launch(void* const* d_in, const int* in_sizes, int n_in,
                              void* d_out, int out_size, void* d_ws, size_t ws_size,
                              hipStream_t stream) {
    const float*    x      = (const float*)d_in[0];
    const int*      packed = (const int*)d_in[1];
    const float*    scales = (const float*)d_in[2];
    const _Float16* ovals  = (const _Float16*)d_in[3];
    const int*      oidx   = (const int*)d_in[4];
    // d_in[5] (ortho_ptr) is uniform KPR stride; row derived arithmetically.
    float* out = (float*)d_out;

    _Float16* Wt = (_Float16*)d_ws;                                      // 32 MB, [in][out]
    _Float16* Xh = (_Float16*)((char*)d_ws + (size_t)OUT_F * IN_F * 2);  // 32 MB
    const int nnz = in_sizes[3];

    {   // 1) dequant + transpose: 64x64 tiles
        dim3 grid(OUT_F / TD, IN_F / TD);
        dequant_wt_k<<<grid, 256, 0, stream>>>(packed, scales, Wt);
    }
    {   // 2) CSR scatter-add into Wt
        csr_add_k<<<(nnz + 255) / 256, 256, 0, stream>>>(ovals, oidx, Wt, nnz);
    }
    {   // 3) x -> f16
        int threads = NTOK * IN_F / 8;
        convert_x_k<<<threads / 256, 256, 0, stream>>>(x, Xh);
    }
    {   // 4) WMMA GEMM
        dim3 grid(OUT_F / BN, NTOK / BM);
        gemm_wmma_k<<<grid, 256, 0, stream>>>(Xh, Wt, out);
    }
}
